// pLoss_all_21517786153438
// MI455X (gfx1250) — matrix operand order repaired and checked by reading, not compiled, not run on hardware
//
#include <hip/hip_runtime.h>

typedef __attribute__((ext_vector_type(16))) _Float16 v16h;
typedef __attribute__((ext_vector_type(2)))  _Float16 v2h;
typedef __attribute__((ext_vector_type(8)))  float    v8f;

constexpr int kB  = 8192;   // batch
constexpr int kL  = 64;     // labels
constexpr int kNS = 4096;   // legal states
constexpr int kBT = 16;     // batch tile per workgroup
constexpr int kWaves = 4;   // waves per workgroup (wave32)
constexpr int kGridMain = kB / kBT;           // 512 workgroups
constexpr int kIters = kNS / (kWaves * 32);   // 32 chunks of 32 states per wave

constexpr float kLog2e = 1.44269504088896340736f;

union U16h { uint4 q[2]; unsigned u[8]; v16h v; };

// lane-half swap (lane ^ 16) via ds_swizzle group-of-32: xor=0x10, and=0x1f
__device__ __forceinline__ float swz16f(float x) {
  return __builtin_bit_cast(float,
      __builtin_amdgcn_ds_swizzle(__builtin_bit_cast(int, x), 0x401f));
}
__device__ __forceinline__ unsigned swz16u(unsigned x) {
  return (unsigned)__builtin_amdgcn_ds_swizzle((int)x, 0x401f);
}

// ---------------------------------------------------------------------------
// Prep: f32 -> f16 copies of S ([NS,L]), S^T ([L,NS]), f ([B,L]) into workspace
// ---------------------------------------------------------------------------
__global__ __launch_bounds__(256) void ploss_prep(
    const float* __restrict__ f, const float* __restrict__ S,
    _Float16* __restrict__ Sh, _Float16* __restrict__ ShT,
    _Float16* __restrict__ fh)
{
  int i = blockIdx.x * 256 + threadIdx.x;
  if (i < kNS * kL) {
    float v = S[i];
    _Float16 h = (_Float16)v;
    Sh[i] = h;
    int s = i >> 6, l = i & 63;
    ShT[(size_t)l * kNS + s] = h;
  }
  if (i < kB * kL) {
    fh[i] = (_Float16)f[i];
  }
}

// ---------------------------------------------------------------------------
// Main fused kernel: GEMM-1 (S@f^T) -> online exp-normalize (log2 domain)
// -> GEMM-2 (S^T@J, plus ones-row WMMA for z) -> pMargin + BCE epilogue.
// One workgroup = 4 waves = 16 batch columns.
// ---------------------------------------------------------------------------
__global__ __launch_bounds__(128) void ploss_main(
    const _Float16* __restrict__ Sh,    // [NS, 64] f16
    const _Float16* __restrict__ ShT,   // [64, NS] f16
    const _Float16* __restrict__ fh,    // [B, 64]  f16
    const float* __restrict__ y,
    const float* __restrict__ msk,
    float* __restrict__ out,            // [0]=loss (finalize), [1..]=pMargin
    float* __restrict__ lossPartial)    // [kGridMain]
{
  __shared__ float s_m[kWaves][kBT];    // per-wave running max (log2 domain)
  __shared__ float s_z[kWaves][kBT];
  __shared__ float s_acc[kL][kBT];
  __shared__ float s_w[kWaves];

  const int tid  = threadIdx.x;
  const int wave = tid >> 5;
  const int lane = tid & 31;
  const int ln   = lane & 15;   // batch column within tile (N), also M for A ops
  const int hl   = lane >> 4;   // lane-half selects K sub-range per ISA layouts
  const int b0   = blockIdx.x * kBT;

  for (int i = tid; i < kL * kBT; i += 128) (&s_acc[0][0])[i] = 0.0f;
  __syncthreads();

  // GEMM-1 B operand: f tile, [K=64 labels x N=16 batch], two K=32 slabs.
  // B layout: lane=N; lanes<16 K={0..7,16..23}, lanes>=16 +8 (matches A).
  const _Float16* fb = fh + (size_t)(b0 + ln) * kL + hl * 8;
  U16h B0, B1;
  B0.q[0] = *(const uint4*)(fb +  0);
  B0.q[1] = *(const uint4*)(fb + 16);
  B1.q[0] = *(const uint4*)(fb + 32);
  B1.q[1] = *(const uint4*)(fb + 48);

  // Constant all-ones A tile: folds z = sum_s J[s,b] into the XDL pipe.
  U16h ONES;
#pragma unroll
  for (int j = 0; j < 8; ++j) ONES.u[j] = 0x3C003C00u;   // {1.0h, 1.0h}

  v8f acc[5];   // [0..3]: GEMM-2 label tiles; [4]: z tile (all rows == z[b])
#pragma unroll
  for (int t = 0; t < 5; ++t)
#pragma unroll
    for (int i = 0; i < 8; ++i) acc[t][i] = 0.0f;

  float m2 = -__builtin_inff();     // running max * log2e for batch col ln

  for (int it = 0; it < kIters; ++it) {
    const int sc = it * (kWaves * 32) + wave * 32;   // this wave's 32 states

    // ---- issue GEMM-1 A loads first (consumed first) ----
    // A layout: lane=M(state); lanes<16 K={0..7,16..23}(+base), lanes>=16 +8
    const _Float16* ar0 = Sh + (size_t)(sc + ln) * kL + hl * 8;
    const _Float16* ar1 = ar0 + 16 * kL;
    U16h A0lo, A0hi, A1lo, A1hi;
    A0lo.q[0] = *(const uint4*)(ar0 +  0);
    A0lo.q[1] = *(const uint4*)(ar0 + 16);
    A0hi.q[0] = *(const uint4*)(ar0 + 32);
    A0hi.q[1] = *(const uint4*)(ar0 + 48);
    A1lo.q[0] = *(const uint4*)(ar1 +  0);
    A1lo.q[1] = *(const uint4*)(ar1 + 16);
    A1hi.q[0] = *(const uint4*)(ar1 + 32);
    A1hi.q[1] = *(const uint4*)(ar1 + 48);

    // ---- issue GEMM-2 A loads now; they land under GEMM-1/exp work ----
    U16h A2t[4];
#pragma unroll
    for (int t = 0; t < 4; ++t) {
      const _Float16* a2 = ShT + (size_t)(t * 16 + ln) * kNS + sc + hl * 16;
      A2t[t].q[0] = *(const uint4*)(a2 +  0);
      A2t[t].q[1] = *(const uint4*)(a2 +  8);
    }

    if (it + 1 < kIters) {
      const int sn = (it + 1) * (kWaves * 32) + wave * 32;
      __builtin_prefetch(Sh + (size_t)(sn + lane) * kL, 0, 1);
    }

    // ---- GEMM-1: two 16-state potential tiles, K=64 via two WMMAs each ----
    v8f c0, c1;
#pragma unroll
    for (int i = 0; i < 8; ++i) { c0[i] = 0.0f; c1[i] = 0.0f; }
    c0 = __builtin_amdgcn_wmma_f32_16x16x32_f16(false, A0lo.v, false, B0.v,
                                                (short)0, c0, false, false);
    c1 = __builtin_amdgcn_wmma_f32_16x16x32_f16(false, A1lo.v, false, B0.v,
                                                (short)0, c1, false, false);
    c0 = __builtin_amdgcn_wmma_f32_16x16x32_f16(false, A0hi.v, false, B1.v,
                                                (short)0, c0, false, false);
    c1 = __builtin_amdgcn_wmma_f32_16x16x32_f16(false, A1hi.v, false, B1.v,
                                                (short)0, c1, false, false);

    // ---- local (per-lane) chunk max, balanced tree ----
    float ma = fmaxf(fmaxf(c0[0], c0[1]), fmaxf(c0[2], c0[3]));
    float mb = fmaxf(fmaxf(c0[4], c0[5]), fmaxf(c0[6], c0[7]));
    float mc = fmaxf(fmaxf(c1[0], c1[1]), fmaxf(c1[2], c1[3]));
    float md = fmaxf(fmaxf(c1[4], c1[5]), fmaxf(c1[6], c1[7]));
    const float cml = fmaxf(fmaxf(ma, mb), fmaxf(mc, md));   // own 16 states

    // Branch test needs no cross-half reduce: __any spans all 32 lanes.
    if (__any(cml * kLog2e > m2)) {              // wave-uniform: EXEC stays full
      const float cm   = fmaxf(cml, swz16f(cml));  // full chunk max, symmetric
      const float mn2  = fmaxf(m2, cm * kLog2e);
      const float resc = __builtin_amdgcn_exp2f(m2 - mn2);  // 0 on first chunk
      m2 = mn2;
#pragma unroll
      for (int t = 0; t < 5; ++t)
#pragma unroll
        for (int i = 0; i < 8; ++i) acc[t][i] *= resc;
    }

    // ---- J = 2^(pot*log2e - m2), pack to f16 (z comes from WMMA below) ----
    float e0[8], e1[8];
#pragma unroll
    for (int i = 0; i < 8; ++i) {
      e0[i] = __builtin_amdgcn_exp2f(__builtin_fmaf(c0[i], kLog2e, -m2));
      e1[i] = __builtin_amdgcn_exp2f(__builtin_fmaf(c1[i], kLog2e, -m2));
    }

    unsigned p0[4], p1[4];
#pragma unroll
    for (int j = 0; j < 4; ++j) {
      p0[j] = __builtin_bit_cast(unsigned,
                __builtin_amdgcn_cvt_pkrtz(e0[2 * j], e0[2 * j + 1]));
      p1[j] = __builtin_bit_cast(unsigned,
                __builtin_amdgcn_cvt_pkrtz(e1[2 * j], e1[2 * j + 1]));
    }
    // Rearrange into B-matrix layout [K=32 states x N=16 batch]:
    //   lanes<16 : V0-3 = own tile0 states 0..7, V4-7 = partner tile0 8..15
    //   lanes>=16: V0-3 = partner tile1 0..7,    V4-7 = own tile1 8..15
    U16h JB;
#pragma unroll
    for (int j = 0; j < 4; ++j) {
      const unsigned pre = hl ? p0[j] : p1[j];   // what my partner needs
      const unsigned got = swz16u(pre);          // what I need from partner
      JB.u[j]     = hl ? got   : p0[j];
      JB.u[4 + j] = hl ? p1[j] : got;
    }

    // ---- GEMM-2 rank-32 update + z via all-ones A tile ----
#pragma unroll
    for (int t = 0; t < 4; ++t)
      acc[t] = __builtin_amdgcn_wmma_f32_16x16x32_f16(false, A2t[t].v, false, JB.v,
                                                      (short)0, acc[t], false, false);
    acc[4] = __builtin_amdgcn_wmma_f32_16x16x32_f16(false, ONES.v, false, JB.v,
                                                    (short)0, acc[4], false, false);
  }

  // ---- cross-wave combine: global max (log2), rescale, reduce into LDS ----
  if (hl == 0) s_m[wave][ln] = m2;
  __syncthreads();
  const float M2 = fmaxf(fmaxf(s_m[0][ln], s_m[1][ln]),
                         fmaxf(s_m[2][ln], s_m[3][ln]));
  const float sca = __builtin_amdgcn_exp2f(m2 - M2);
  // acc[4] rows are identical == wave-local z[b] (K spanned both lane halves)
  if (hl == 0) s_z[wave][ln] = acc[4][0] * sca;
#pragma unroll
  for (int t = 0; t < 4; ++t)
#pragma unroll
    for (int i = 0; i < 8; ++i)
      atomicAdd(&s_acc[t * 16 + i + 8 * hl][ln], acc[t][i] * sca);
  __syncthreads();

  // ---- epilogue: pMargin write + BCE partial ----
  float lpart = 0.0f;
#pragma unroll
  for (int k = 0; k < 8; ++k) {
    const int flat = tid + 128 * k;          // 0..1023
    const int b = flat >> 6, l = flat & 63;
    const float zb = s_z[0][b] + s_z[1][b] + s_z[2][b] + s_z[3][b];
    const float pm = s_acc[l][b] / zb;
    const size_t o = (size_t)(b0 + b) * kL + l;
    out[1 + o] = pm;
    const float yy = y[o], mk = msk[o];
    const float lp = fmaxf(__logf(pm), -100.0f);
    const float l1 = fmaxf(__logf(1.0f - pm), -100.0f);
    lpart += -(yy * lp + (1.0f - yy) * l1) * mk;
  }
#pragma unroll
  for (int off = 16; off; off >>= 1) lpart += __shfl_xor(lpart, off, 32);
  if (lane == 0) s_w[wave] = lpart;
  __syncthreads();
  if (tid == 0)
    lossPartial[blockIdx.x] = s_w[0] + s_w[1] + s_w[2] + s_w[3];
}

// ---------------------------------------------------------------------------
// Deterministic finalize: fixed-order sum of 512 partials -> loss scalar
// ---------------------------------------------------------------------------
__global__ void ploss_finalize(const float* __restrict__ partials,
                               float* __restrict__ out)
{
  if (threadIdx.x == 0 && blockIdx.x == 0) {
    float s = 0.0f;
    for (int i = 0; i < kGridMain; ++i) s += partials[i];
    out[0] = s * (1.0f / (float)(kB * kL));
  }
}

extern "C" void kernel_launch(void* const* d_in, const int* in_sizes, int n_in,
                              void* d_out, int out_size, void* d_ws, size_t ws_size,
                              hipStream_t stream)
{
  (void)in_sizes; (void)n_in; (void)out_size; (void)ws_size;
  const float* f   = (const float*)d_in[0];
  const float* yv  = (const float*)d_in[1];
  const float* mk  = (const float*)d_in[2];
  const float* S   = (const float*)d_in[3];

  char* ws = (char*)d_ws;
  _Float16* Sh  = (_Float16*)(ws);                                  // 512 KB
  _Float16* ShT = (_Float16*)(ws + (size_t)kNS * kL * 2);           // 512 KB
  _Float16* fh  = (_Float16*)(ws + (size_t)kNS * kL * 4);           // 1 MB
  float* partials = (float*)(ws + (size_t)kNS * kL * 4 + (size_t)kB * kL * 2);
  float* out = (float*)d_out;

  ploss_prep<<<(kB * kL) / 256, 256, 0, stream>>>(f, S, Sh, ShT, fh);
  ploss_main<<<kGridMain, 128, 0, stream>>>(Sh, ShT, fh, yv, mk, out, partials);
  ploss_finalize<<<1, 32, 0, stream>>>(partials, out);
}
